// StateModelEncoder_60507499266592
// MI455X (gfx1250) — compile-verified, matrix operand order checked by reading, not run on hardware
//
#include <hip/hip_runtime.h>
#include <hip/hip_bf16.h>
#include <math.h>

// ---------------------------------------------------------------------------
// Problem constants (match reference)
// ---------------------------------------------------------------------------
#define NV   100000
#define NS   50000
#define E_VV 1000000
#define E_VS 500000
#define E_SS 500000
#define HDIM 64

typedef __attribute__((ext_vector_type(16))) _Float16 v16h;
typedef __attribute__((ext_vector_type(8)))  float    v8f;

// ---------------------------------------------------------------------------
// WMMA GEMM:  Y[n,64] = op( X[n,din] @ W[din,64] + bias [+ Y] ) , op = relu?
// One wave holds all B (weight) operands in registers and grid-strides over
// 16-row tiles, so weight loads are amortized over ~4 tiles (32 WMMAs).
// A operand (16x32 f16): lane%16 = M row; halves h<8 -> K = 8g+h (contiguous),
// halves h>=8 -> K = 16+8g+(h-8) (contiguous), g = lane/16 (ISA 7.12.2).
// B operand (32x16): lane%16 = N col, same K map. C/D: N = lane%16, M = r+8g.
// ---------------------------------------------------------------------------
template<int DIN, bool BIAS, bool ACCUM, bool RELU>
__global__ __launch_bounds__(256)
void gemm_wmma(const float* __restrict__ X, int ldx,
               const float* __restrict__ W,      // [DIN,64] row-major
               const float* __restrict__ bias,   // [64] (may be null if !BIAS)
               float* __restrict__ Y,            // [n,64]
               int nrows) {
  constexpr int KTILES = (DIN + 31) / 32;
  const int lane   = threadIdx.x & 31;
  const int g      = lane >> 4;
  const int l16    = lane & 15;
  const int waveId = blockIdx.x * (blockDim.x >> 5) + (threadIdx.x >> 5);
  const int nWaves = gridDim.x * (blockDim.x >> 5);
  const int ntiles = nrows >> 4;

  // ---- hoist B operands + bias: loaded once, reused across all row tiles ----
  v16h  b[4][KTILES];
  float bv[4];
#pragma unroll
  for (int nt = 0; nt < 4; ++nt) {
    const int colN = nt * 16 + l16;
    bv[nt] = BIAS ? bias[colN] : 0.0f;
#pragma unroll
    for (int kt = 0; kt < KTILES; ++kt) {
      v16h t;
#pragma unroll
      for (int h = 0; h < 16; ++h) {
        int K = kt * 32 + ((h < 8) ? (g * 8 + h) : (8 + g * 8 + h));
        if (DIN == 64) {
          t[h] = (_Float16)W[(size_t)K * HDIM + colN];
        } else {           // zero-pad: unconditional clamped load + value select
          int Kc = (K < DIN) ? K : 0;
          float w = W[(size_t)Kc * HDIM + colN];
          t[h] = (_Float16)((K < DIN) ? w : 0.0f);
        }
      }
      b[nt][kt] = t;
    }
  }

  // ---- grid-stride over 16-row tiles (loop bound is wave-uniform) ----
  for (int tile = waveId; tile < ntiles; tile += nWaves) {
    const int rowBase = tile << 4;
    const int row     = rowBase + l16;

    v16h a[KTILES];
#pragma unroll
    for (int kt = 0; kt < KTILES; ++kt) {
      v16h av;
      if (DIN == 64) {     // two contiguous 8-float runs per lane -> b128 loads
        const float* base = X + (size_t)row * ldx + kt * 32 + g * 8;
        float4 x0 = *(const float4*)(base);
        float4 x1 = *(const float4*)(base + 4);
        float4 x2 = *(const float4*)(base + 16);
        float4 x3 = *(const float4*)(base + 20);
        av[0]=(_Float16)x0.x; av[1]=(_Float16)x0.y; av[2]=(_Float16)x0.z; av[3]=(_Float16)x0.w;
        av[4]=(_Float16)x1.x; av[5]=(_Float16)x1.y; av[6]=(_Float16)x1.z; av[7]=(_Float16)x1.w;
        av[8]=(_Float16)x2.x; av[9]=(_Float16)x2.y; av[10]=(_Float16)x2.z; av[11]=(_Float16)x2.w;
        av[12]=(_Float16)x3.x; av[13]=(_Float16)x3.y; av[14]=(_Float16)x3.z; av[15]=(_Float16)x3.w;
      } else {
#pragma unroll
        for (int h = 0; h < 16; ++h) {
          int K = (h < 8) ? (g * 8 + h) : (8 + g * 8 + h);
          int Kc = (K < DIN) ? K : 0;
          float x = X[(size_t)row * ldx + Kc];
          av[h] = (_Float16)((K < DIN) ? x : 0.0f);
        }
      }
      a[kt] = av;
    }

#pragma unroll
    for (int nt = 0; nt < 4; ++nt) {
      const int colN = nt * 16 + l16;
      v8f c;
#pragma unroll
      for (int r = 0; r < 8; ++r) {
        float v = bv[nt];
        if (ACCUM) v += Y[(size_t)(rowBase + g * 8 + r) * HDIM + colN];
        c[r] = v;
      }
#pragma unroll
      for (int kt = 0; kt < KTILES; ++kt)
        c = __builtin_amdgcn_wmma_f32_16x16x32_f16(false, a[kt], false, b[nt][kt],
                                                   (short)0, c, false, false);
#pragma unroll
      for (int r = 0; r < 8; ++r) {
        float v = c[r];
        if (RELU) v = v > 0.0f ? v : 0.0f;
        Y[(size_t)(rowBase + g * 8 + r) * HDIM + colN] = v;
      }
    }
  }
}

// ---------------------------------------------------------------------------
// Elementwise / scatter kernels (memory-bound part; f32 atomics resolve in L2)
// ---------------------------------------------------------------------------
__global__ void zero_f32(float* __restrict__ p, int n) {
  int i4 = blockIdx.x * blockDim.x + threadIdx.x;
  int n4 = n >> 2;
  float4 z = make_float4(0.f, 0.f, 0.f, 0.f);
  for (; i4 < n4; i4 += gridDim.x * blockDim.x)
    ((float4*)p)[i4] = z;
}

__global__ void count_dst(const int* __restrict__ col, int E, float* __restrict__ cnt) {
  int e = blockIdx.x * blockDim.x + threadIdx.x;
  if (e < E) atomicAdd(&cnt[col[e]], 1.0f);
}

__global__ void deg_rsqrt(float* __restrict__ d, int n) {
  int i = blockIdx.x * blockDim.x + threadIdx.x;
  if (i < n) { float v = d[i]; d[i] = (v > 0.0f) ? rsqrtf(v) : 0.0f; }
}

// TAG hop, 7-wide features (src stride lds, dst padded stride 8)
__global__ void hop7(const int* __restrict__ row, const int* __restrict__ col,
                     const float* __restrict__ dinv, const float* __restrict__ src,
                     int lds, float* __restrict__ dst, int E) {
  int e = blockIdx.x * blockDim.x + threadIdx.x;
  if (e >= E) return;
  int r = row[e], c = col[e];
  float nrm = dinv[r] * dinv[c];
#pragma unroll
  for (int f = 0; f < 7; ++f)
    atomicAdd(&dst[(size_t)c * 8 + f], nrm * src[(size_t)r * lds + f]);
}

// TAG hop, 64-wide features: one thread per (edge, feature)
__global__ void hop64(const int* __restrict__ row, const int* __restrict__ col,
                      const float* __restrict__ dinv, const float* __restrict__ src,
                      float* __restrict__ dst, int E) {
  int t = blockIdx.x * blockDim.x + threadIdx.x;
  int e = t >> 6, f = t & 63;
  if (e >= E) return;
  int r = row[e], c = col[e];
  float nrm = dinv[r] * dinv[c];
  atomicAdd(&dst[(size_t)c * HDIM + f], nrm * src[(size_t)r * HDIM + f]);
}

// RGCN: per-relation masked sum + count
__global__ void scatter_rel(const int* __restrict__ row, const int* __restrict__ col,
                            const int* __restrict__ et, int rel,
                            const float* __restrict__ x, float* __restrict__ acc,
                            float* __restrict__ cnt, int E) {
  int t = blockIdx.x * blockDim.x + threadIdx.x;
  int e = t >> 6, f = t & 63;
  if (e >= E || et[e] != rel) return;
  int r = row[e], c = col[e];
  atomicAdd(&acc[(size_t)c * HDIM + f], x[(size_t)r * HDIM + f]);
  if (f == 0) atomicAdd(&cnt[c], 1.0f);
}

// SAGE: unmasked sum + count
__global__ void scatter_mean(const int* __restrict__ row, const int* __restrict__ col,
                             const float* __restrict__ x, float* __restrict__ acc,
                             float* __restrict__ cnt, int E) {
  int t = blockIdx.x * blockDim.x + threadIdx.x;
  int e = t >> 6, f = t & 63;
  if (e >= E) return;
  int r = row[e], c = col[e];
  atomicAdd(&acc[(size_t)c * HDIM + f], x[(size_t)r * HDIM + f]);
  if (f == 0) atomicAdd(&cnt[c], 1.0f);
}

__global__ void divide_cnt(float* __restrict__ acc, const float* __restrict__ cnt, int n) {
  int t = blockIdx.x * blockDim.x + threadIdx.x;
  if (t >= n * HDIM) return;
  acc[t] /= fmaxf(cnt[t >> 6], 1.0f);
}

// ResGated: gate = sigmoid(k[dst] + q[src] + edge_attr@We + be); agg += gate*v[src]
__global__ void gate_scatter(const int* __restrict__ row, const int* __restrict__ col,
                             const float* __restrict__ ea,   // [E,2]
                             const float* __restrict__ kk,   // [NS,64]
                             const float* __restrict__ qq,   // [NV,64]
                             const float* __restrict__ vv,   // [NV,64]
                             const float* __restrict__ We,   // [2,64]
                             const float* __restrict__ be,   // [64]
                             float* __restrict__ agg, int E) {
  int t = blockIdx.x * blockDim.x + threadIdx.x;
  int e = t >> 6, f = t & 63;
  if (e >= E) return;
  int r = row[e], c = col[e];
  float ep = ea[(size_t)e * 2] * We[f] + ea[(size_t)e * 2 + 1] * We[HDIM + f] + be[f];
  float gsum = kk[(size_t)c * HDIM + f] + qq[(size_t)r * HDIM + f] + ep;
  float gate = 1.0f / (1.0f + expf(-gsum));
  atomicAdd(&agg[(size_t)c * HDIM + f], gate * vv[(size_t)r * HDIM + f]);
}

// Final linear 64 -> 1
__global__ void final_lin(const float* __restrict__ x, const float* __restrict__ w,
                          const float* __restrict__ b, float* __restrict__ out, int n) {
  int i = blockIdx.x * blockDim.x + threadIdx.x;
  if (i >= n) return;
  float acc = b[0];
#pragma unroll
  for (int f = 0; f < HDIM; ++f) acc += x[(size_t)i * HDIM + f] * w[f];
  out[i] = acc;
}

// ---------------------------------------------------------------------------
// Host-side orchestration
// ---------------------------------------------------------------------------
static inline int gb(long long n, int t) { return (int)((n + t - 1) / t); }

extern "C" void kernel_launch(void* const* d_in, const int* in_sizes, int n_in,
                              void* d_out, int out_size, void* d_ws, size_t ws_size,
                              hipStream_t stream) {
  // --- inputs (setup_inputs order) ---
  const float* game_x   = (const float*)d_in[0];            // [NV,7]
  const float* state_x  = (const float*)d_in[1];            // [NS,7]
  const int*   evv      = (const int*)d_in[2];              // [2,E_VV]
  const int*   et_vv    = (const int*)d_in[3];              // [E_VV]
  const int*   ehist    = (const int*)d_in[4];              // [2,E_VS]
  const float* attrhist = (const float*)d_in[5];            // [E_VS,2]
  const int*   ein      = (const int*)d_in[6];              // [2,E_VS]
  const int*   ess      = (const int*)d_in[7];              // [2,E_SS]
  // --- params (dict insertion order) ---
  int pi = 8;
  const float* tag10_W   = (const float*)d_in[pi++]; // [4,7,64]
  const float* tag10_b   = (const float*)d_in[pi++];
  const float* tag2_W    = (const float*)d_in[pi++]; // [4,64,64]
  const float* tag2_b    = (const float*)d_in[pi++];
  const float* rgcn_W    = (const float*)d_in[pi++]; // [3,64,64]
  const float* rgcn_root = (const float*)d_in[pi++];
  const float* rgcn_b    = (const float*)d_in[pi++];
  const float* rg_Wk  = (const float*)d_in[pi++]; const float* rg_bk = (const float*)d_in[pi++];
  const float* rg_Wq  = (const float*)d_in[pi++]; const float* rg_bq = (const float*)d_in[pi++];
  const float* rg_Wv  = (const float*)d_in[pi++]; const float* rg_bv = (const float*)d_in[pi++];
  const float* rg_We  = (const float*)d_in[pi++]; const float* rg_be = (const float*)d_in[pi++];
  const float* rg_Wskip = (const float*)d_in[pi++];
  const float* rg_bias  = (const float*)d_in[pi++];
  const float* s32_Wl = (const float*)d_in[pi++]; const float* s32_bl = (const float*)d_in[pi++];
  const float* s32_Wr = (const float*)d_in[pi++];
  const float* s4_Wl  = (const float*)d_in[pi++]; const float* s4_bl  = (const float*)d_in[pi++];
  const float* s4_Wr  = (const float*)d_in[pi++];
  const float* s42_Wl = (const float*)d_in[pi++]; const float* s42_bl = (const float*)d_in[pi++];
  const float* s42_Wr = (const float*)d_in[pi++];
  const float* s5_Wl  = (const float*)d_in[pi++]; const float* s5_bl  = (const float*)d_in[pi++];
  const float* s5_Wr  = (const float*)d_in[pi++];
  const float* lin_W  = (const float*)d_in[pi++];
  const float* lin_b  = (const float*)d_in[pi++];

  // edge index rows/cols
  const int *evv_r = evv, *evv_c = evv + E_VV;
  const int *eh_r = ehist, *eh_c = ehist + E_VS;
  const int *ei_r = ein,   *ei_c = ein   + E_VS;
  const int *es_r = ess,   *es_c = ess   + E_SS;

  // --- workspace layout (floats) ---
  float* ws = (float*)d_ws;
  const size_t VSZ = (size_t)NV * HDIM, SSZ = (size_t)NS * HDIM;
  float* V0 = ws;                  // [NV,64]
  float* V1 = V0 + VSZ;            // [NV,64]
  float* V2 = V1 + VSZ;            // [NV,64]
  float* S0 = V2 + VSZ;            // [NS,64]
  float* S1 = S0 + SSZ;            // [NS,64]
  float* S2 = S1 + SSZ;            // [NS,64]
  float* H7A = S2 + SSZ;           // [NV,8]
  float* H7B = H7A + (size_t)NV * 8;
  float* DEG = H7B + (size_t)NV * 8;  // [NV] (also dinv)
  float* CNT = DEG + NV;              // [NV]

  const dim3 T(256);
  auto Z = [&](float* p, long long n){
    int blocks = gb(n / 4, 256); if (blocks > 4096) blocks = 4096;
    zero_f32<<<blocks, T, 0, stream>>>(p, (int)n);
  };
  // GEMM grids: 8 waves/block, ~4 row tiles per wave (weights stay in regs)
  const int gv  = gb((long long)(NV >> 4), 8 * 4);   // 196 blocks for NV
  const int gs  = gb((long long)(NS >> 4), 8 * 4);   // 98 blocks for NS
  const int eVV64 = gb((long long)E_VV * 64, 256);
  const int eVS64 = gb((long long)E_VS * 64, 256);
  const int eSS64 = gb((long long)E_SS * 64, 256);

  // ======================= TAGConv(game_x, e_vv) ==========================
  Z(DEG, NV);
  count_dst<<<gb(E_VV,256), T, 0, stream>>>(evv_c, E_VV, DEG);
  deg_rsqrt<<<gb(NV,256), T, 0, stream>>>(DEG, NV);
  gemm_wmma<7,true,false,false><<<gv, T, 0, stream>>>(game_x, 7, tag10_W, tag10_b, V0, NV);
  Z(H7A, (long long)NV*8);
  hop7<<<gb(E_VV,256), T, 0, stream>>>(evv_r, evv_c, DEG, game_x, 7, H7A, E_VV);
  gemm_wmma<7,false,true,false><<<gv, T, 0, stream>>>(H7A, 8, tag10_W + 1*7*64, nullptr, V0, NV);
  Z(H7B, (long long)NV*8);
  hop7<<<gb(E_VV,256), T, 0, stream>>>(evv_r, evv_c, DEG, H7A, 8, H7B, E_VV);
  gemm_wmma<7,false,true,false><<<gv, T, 0, stream>>>(H7B, 8, tag10_W + 2*7*64, nullptr, V0, NV);
  Z(H7A, (long long)NV*8);
  hop7<<<gb(E_VV,256), T, 0, stream>>>(evv_r, evv_c, DEG, H7B, 8, H7A, E_VV);
  gemm_wmma<7,false,true,true><<<gv, T, 0, stream>>>(H7A, 8, tag10_W + 3*7*64, nullptr, V0, NV);
  // g1 = relu(tag10) in V0

  // ======================= RGCNConv(g1, e_vv, et) =========================
  gemm_wmma<64,true,false,false><<<gv, T, 0, stream>>>(V0, 64, rgcn_root, rgcn_b, V1, NV);
  for (int r = 0; r < 3; ++r) {
    Z(V2, (long long)VSZ); Z(CNT, NV);
    scatter_rel<<<eVV64, T, 0, stream>>>(evv_r, evv_c, et_vv, r, V0, V2, CNT, E_VV);
    divide_cnt<<<gb((long long)VSZ,256), T, 0, stream>>>(V2, CNT, NV);
    if (r == 2) gemm_wmma<64,false,true,true ><<<gv, T, 0, stream>>>(V2, 64, rgcn_W + (size_t)r*64*64, nullptr, V1, NV);
    else        gemm_wmma<64,false,true,false><<<gv, T, 0, stream>>>(V2, 64, rgcn_W + (size_t)r*64*64, nullptr, V1, NV);
  }
  // g2 = relu(rgcn) in V1

  // =================== ResGatedGraphConv(g2 -> state) =====================
  gemm_wmma<7 ,true,false,false><<<gs, T, 0, stream>>>(state_x, 7, rg_Wk, rg_bk, S0, NS); // k
  gemm_wmma<64,true,false,false><<<gv, T, 0, stream>>>(V1, 64, rg_Wq, rg_bq, V0, NV);     // q (g1 dead)
  gemm_wmma<64,true,false,false><<<gv, T, 0, stream>>>(V1, 64, rg_Wv, rg_bv, V2, NV);     // v
  Z(S1, (long long)SSZ);
  gate_scatter<<<eVS64, T, 0, stream>>>(eh_r, eh_c, attrhist, S0, V0, V2, rg_We, rg_be, S1, E_VS);
  gemm_wmma<7,true,true,true><<<gs, T, 0, stream>>>(state_x, 7, rg_Wskip, rg_bias, S1, NS);
  // s1 = relu(resgated) in S1

  // ========================= SAGE s32 (e_hist) ============================
  Z(S0, (long long)SSZ); Z(CNT, NS);
  scatter_mean<<<eVS64, T, 0, stream>>>(eh_r, eh_c, V1, S0, CNT, E_VS);
  divide_cnt<<<gb((long long)SSZ,256), T, 0, stream>>>(S0, CNT, NS);
  gemm_wmma<64,true,false,false><<<gs, T, 0, stream>>>(S0, 64, s32_Wl, s32_bl, S2, NS);
  gemm_wmma<64,false,true,true ><<<gs, T, 0, stream>>>(S1, 64, s32_Wr, nullptr, S2, NS);
  // s2 in S2

  // ====================== SAGE s4 + s42 (e_in) ============================
  Z(S0, (long long)SSZ); Z(CNT, NS);
  scatter_mean<<<eVS64, T, 0, stream>>>(ei_r, ei_c, V1, S0, CNT, E_VS);
  divide_cnt<<<gb((long long)SSZ,256), T, 0, stream>>>(S0, CNT, NS);
  gemm_wmma<64,true,false,false><<<gs, T, 0, stream>>>(S0, 64, s4_Wl, s4_bl, S1, NS);
  gemm_wmma<64,false,true,true ><<<gs, T, 0, stream>>>(S2, 64, s4_Wr, nullptr, S1, NS);   // s3 in S1
  // s42 reuses the same mean (same x_src = g2, same edges)
  gemm_wmma<64,true,false,false><<<gs, T, 0, stream>>>(S0, 64, s42_Wl, s42_bl, S2, NS);
  gemm_wmma<64,false,true,true ><<<gs, T, 0, stream>>>(S1, 64, s42_Wr, nullptr, S2, NS);  // s4out in S2

  // ========================= TAGConv tag2 (e_ss) ==========================
  Z(DEG, NS);
  count_dst<<<gb(E_SS,256), T, 0, stream>>>(es_c, E_SS, DEG);
  deg_rsqrt<<<gb(NS,256), T, 0, stream>>>(DEG, NS);
  gemm_wmma<64,true,false,false><<<gs, T, 0, stream>>>(S2, 64, tag2_W, tag2_b, S1, NS);
  Z(S0, (long long)SSZ);
  hop64<<<eSS64, T, 0, stream>>>(es_r, es_c, DEG, S2, S0, E_SS);
  gemm_wmma<64,false,true,false><<<gs, T, 0, stream>>>(S0, 64, tag2_W + 1*64*64, nullptr, S1, NS);
  Z(S2, (long long)SSZ);
  hop64<<<eSS64, T, 0, stream>>>(es_r, es_c, DEG, S0, S2, E_SS);
  gemm_wmma<64,false,true,false><<<gs, T, 0, stream>>>(S2, 64, tag2_W + 2*64*64, nullptr, S1, NS);
  Z(S0, (long long)SSZ);
  hop64<<<eSS64, T, 0, stream>>>(es_r, es_c, DEG, S2, S0, E_SS);
  gemm_wmma<64,false,true,true ><<<gs, T, 0, stream>>>(S0, 64, tag2_W + 3*64*64, nullptr, S1, NS);
  // s5in = relu(tag2) in S1

  // ========================= SAGE s5 (e_ss) ===============================
  Z(S0, (long long)SSZ); Z(CNT, NS);
  scatter_mean<<<eSS64, T, 0, stream>>>(es_r, es_c, S1, S0, CNT, E_SS);
  divide_cnt<<<gb((long long)SSZ,256), T, 0, stream>>>(S0, CNT, NS);
  gemm_wmma<64,true,false,false><<<gs, T, 0, stream>>>(S0, 64, s5_Wl, s5_bl, S2, NS);
  gemm_wmma<64,false,true,true ><<<gs, T, 0, stream>>>(S1, 64, s5_Wr, nullptr, S2, NS);

  // ============================ Final linear ==============================
  final_lin<<<gb(NS,256), T, 0, stream>>>(S2, lin_W, lin_b, (float*)d_out, NS);
}
